// GAT_Fusion_2147483648587
// MI455X (gfx1250) — matrix operand order repaired and checked by reading, not compiled, not run on hardware
//
#include <hip/hip_runtime.h>

// ---------------------------------------------------------------------------
// Types for CDNA5 WMMA (wave32)
// ---------------------------------------------------------------------------
typedef __bf16 bf16_t;
typedef __attribute__((ext_vector_type(16))) __bf16 v16bf;
typedef __attribute__((ext_vector_type(8)))  float  v8f;

#define NNODES 50000
#define MT     3125          // 50000 / 16, exact
#define NEG_SLOPE 0.2f

// ---------------------------------------------------------------------------
// Pack fp32 row-major [Msrc x Ksrc] into CDNA5 16-bit A-fragment layout.
// Fragment (mtile, ktOff+kt) occupies 512 contiguous bf16; within a fragment,
// element t = lane*16 + i maps to A[m0 + (lane&15)][k0 + klocal] where
// klocal = (i>=8?16:0) + ((lane>>4)&1)*8 + (i&7)   (ISA 7.12.2, 16-bit A 16x32)
// Optional: bias[k] added, act==1 -> ELU.
// ---------------------------------------------------------------------------
__global__ void pack_a_kernel(const float* __restrict__ src, bf16_t* __restrict__ dst,
                              int Msrc, int Ksrc, int KTdst, int ktOff,
                              const float* __restrict__ bias, int act) {
    int KTsrc = Ksrc >> 5;
    int mtile = blockIdx.x / KTsrc;
    int kt    = blockIdx.x % KTsrc;
    int t     = threadIdx.x;          // 0..511
    int lane  = t >> 4;
    int i     = t & 15;
    int m = mtile * 16 + (lane & 15);
    int k = kt * 32 + ((i >= 8) ? 16 : 0) + ((lane >> 4) & 1) * 8 + (i & 7);
    float v = 0.f;
    if (m < Msrc) {
        v = src[(size_t)m * Ksrc + k];
        if (bias) v += bias[k];
        if (act == 1) v = (v > 0.f) ? v : (__expf(v) - 1.f);   // ELU
    }
    dst[((size_t)(mtile * KTdst + ktOff + kt)) * 512 + t] = (bf16_t)v;
}

// ---------------------------------------------------------------------------
// Pack fp32 W [K x Nc] (row-major) into CDNA5 16-bit B-fragment layout.
// Fragment f = nt*KT + kt; element t = lane*16 + i maps to
// B[k0 + (i>=8?16:0) + ((lane>>4)&1)*8 + (i&7)][n0 + (lane&15)]   (same k order as A)
// ---------------------------------------------------------------------------
__global__ void pack_b_kernel(const float* __restrict__ W, bf16_t* __restrict__ dst,
                              int K, int Nc) {
    int KT = K >> 5;
    int kt = blockIdx.x % KT;
    int nt = blockIdx.x / KT;
    int t    = threadIdx.x;
    int lane = t >> 4;
    int i    = t & 15;
    int n = nt * 16 + (lane & 15);
    int k = kt * 32 + ((i >= 8) ? 16 : 0) + ((lane >> 4) & 1) * 8 + (i & 7);
    dst[(size_t)blockIdx.x * 512 + t] = (bf16_t)W[(size_t)k * Nc + n];
}

// ---------------------------------------------------------------------------
// WMMA GEMM: C[MT*16 x Ntiles*16] = A(packed) x B(packed), f32 accumulate.
// One wave per 16x16 output tile; block = Ntiles waves; grid.x = Mtiles.
// Inner loop: two 32B fragment loads + v_wmma_f32_16x16x32_bf16.
// WGP-scope prefetch (locality 3) pulls the next k-tile into all cache levels.
// ---------------------------------------------------------------------------
__global__ void wmma_gemm_kernel(const bf16_t* __restrict__ Apk,
                                 const bf16_t* __restrict__ Bpk,
                                 float* __restrict__ C,
                                 int KT, int ldC) {
    int mtile = blockIdx.x;
    int nt    = threadIdx.x >> 5;
    int lane  = threadIdx.x & 31;
    const bf16_t* aBase = Apk + ((size_t)mtile * KT) * 512 + lane * 16;
    const bf16_t* bBase = Bpk + ((size_t)nt    * KT) * 512 + lane * 16;
    v8f acc = {};
    for (int kt = 0; kt < KT; ++kt) {
        v16bf a = *(const v16bf*)(aBase + (size_t)kt * 512);
        v16bf b = *(const v16bf*)(bBase + (size_t)kt * 512);
        __builtin_prefetch(aBase + (size_t)(kt + 1) * 512, 0, 3);  // WGP scope
        __builtin_prefetch(bBase + (size_t)(kt + 1) * 512, 0, 3);
        acc = __builtin_amdgcn_wmma_f32_16x16x32_bf16(
                  false, a, false, b, (short)0, acc, false, false);
    }
    // D layout: VGPR r -> row r (+8 for lanes 16..31), col = lane&15
    int row = mtile * 16 + ((lane >> 4) & 1) * 8;
    int col = nt * 16 + (lane & 15);
    float* cp = C + (size_t)row * ldC + col;
#pragma unroll
    for (int r = 0; r < 8; ++r) cp[(size_t)r * ldC] = acc[r];
}

// ---------------------------------------------------------------------------
// alpha_s / alpha_d: wave per (node, head); lane = channel (C == 32 == wave32)
// ---------------------------------------------------------------------------
__global__ void alpha_kernel(const float* __restrict__ h,
                             const float* __restrict__ a_s, const float* __restrict__ a_d,
                             float* __restrict__ as, float* __restrict__ ad,
                             int Nn, int H) {
    int wid  = (int)((blockIdx.x * (size_t)blockDim.x + threadIdx.x) >> 5);
    int lane = threadIdx.x & 31;
    int n = wid / H, hh = wid % H;
    if (n >= Nn) return;
    float v = h[(size_t)n * (H * 32) + hh * 32 + lane];
    float s = v * a_s[hh * 32 + lane];
    float d = v * a_d[hh * 32 + lane];
    for (int off = 16; off >= 1; off >>= 1) {
        s += __shfl_xor(s, off, 32);
        d += __shfl_xor(d, off, 32);
    }
    if (lane == 0) { as[(size_t)n * H + hh] = s; ad[(size_t)n * H + hh] = d; }
}

__global__ void fill_kernel(float* p, float val, size_t n) {
    size_t i = blockIdx.x * (size_t)blockDim.x + threadIdx.x;
    if (i < n) p[i] = val;
}

// int-punned float atomic max (correct for mixed signs; init = -inf)
__device__ inline void atomicMaxF(float* addr, float v) {
    if (v >= 0.f) atomicMax((int*)addr, __float_as_int(v));
    else          atomicMin((unsigned int*)addr, (unsigned int)__float_as_int(v));
}

__global__ void edge_max_kernel(const long long* __restrict__ srcI,
                                const long long* __restrict__ dstI,
                                long long E, long long EE,
                                const float* __restrict__ as, const float* __restrict__ ad,
                                float* __restrict__ m, int H) {
    long long e = blockIdx.x * (long long)blockDim.x + threadIdx.x;
    if (e >= EE) return;
    long long s, d;
    if (e < E) { s = srcI[e]; d = dstI[e]; } else { s = e - E; d = s; }
    for (int hh = 0; hh < H; ++hh) {
        float x = as[s * H + hh] + ad[d * H + hh];
        x = (x > 0.f) ? x : NEG_SLOPE * x;
        atomicMaxF(&m[d * H + hh], x);
    }
}

__global__ void edge_exp_kernel(const long long* __restrict__ srcI,
                                const long long* __restrict__ dstI,
                                long long E, long long EE,
                                const float* __restrict__ as, const float* __restrict__ ad,
                                const float* __restrict__ m, float* __restrict__ dn, int H) {
    long long e = blockIdx.x * (long long)blockDim.x + threadIdx.x;
    if (e >= EE) return;
    long long s, d;
    if (e < E) { s = srcI[e]; d = dstI[e]; } else { s = e - E; d = s; }
    for (int hh = 0; hh < H; ++hh) {
        float x = as[s * H + hh] + ad[d * H + hh];
        x = (x > 0.f) ? x : NEG_SLOPE * x;
        atomicAdd(&dn[d * H + hh], __expf(x - m[d * H + hh]));
    }
}

// wave per edge; lane = channel; recompute softmax weight (ALU is free)
__global__ void edge_agg_kernel(const long long* __restrict__ srcI,
                                const long long* __restrict__ dstI,
                                long long E, long long EE,
                                const float* __restrict__ as, const float* __restrict__ ad,
                                const float* __restrict__ m, const float* __restrict__ dn,
                                const float* __restrict__ h, float* __restrict__ agg, int H) {
    long long wid = (blockIdx.x * (long long)blockDim.x + threadIdx.x) >> 5;
    int lane = threadIdx.x & 31;
    if (wid >= EE) return;
    long long s, d;
    if (wid < E) { s = srcI[wid]; d = dstI[wid]; } else { s = wid - E; d = s; }
    int HC = H * 32;
    for (int hh = 0; hh < H; ++hh) {
        float x = as[s * H + hh] + ad[d * H + hh];
        x = (x > 0.f) ? x : NEG_SLOPE * x;
        float w = __expf(x - m[d * H + hh]) / (dn[d * H + hh] + 1e-16f);
        float val = h[(size_t)s * HC + hh * 32 + lane] * w;
        atomicAdd(&agg[(size_t)d * HC + hh * 32 + lane], val);
    }
}

// out[n] = bf2 + sum_c relu(z1[n,c] + bf1[c]) * Wf2[c]   (wave per node, 64 ch)
__global__ void final_kernel(const float* __restrict__ z1, const float* __restrict__ bf1,
                             const float* __restrict__ Wf2, const float* __restrict__ bf2,
                             float* __restrict__ out, int Nn) {
    int wid  = (int)((blockIdx.x * (size_t)blockDim.x + threadIdx.x) >> 5);
    int lane = threadIdx.x & 31;
    if (wid >= Nn) return;
    float sum = 0.f;
#pragma unroll
    for (int c0 = 0; c0 < 64; c0 += 32) {
        int c = c0 + lane;
        float v = z1[(size_t)wid * 64 + c] + bf1[c];
        v = (v > 0.f) ? v : 0.f;
        sum += v * Wf2[c];
    }
    for (int off = 16; off >= 1; off >>= 1) sum += __shfl_xor(sum, off, 32);
    if (lane == 0) out[wid] = sum + bf2[0];
}

// ---------------------------------------------------------------------------
extern "C" void kernel_launch(void* const* d_in, const int* in_sizes, int n_in,
                              void* d_out, int out_size, void* d_ws, size_t ws_size,
                              hipStream_t stream) {
    const float* txt    = (const float*)d_in[0];
    const float* x      = (const float*)d_in[1];
    const float* W1     = (const float*)d_in[2];
    const float* a_src1 = (const float*)d_in[3];
    const float* a_dst1 = (const float*)d_in[4];
    const float* b1     = (const float*)d_in[5];
    const float* W2     = (const float*)d_in[6];
    const float* a_src2 = (const float*)d_in[7];
    const float* a_dst2 = (const float*)d_in[8];
    const float* b2     = (const float*)d_in[9];
    const float* Wf1    = (const float*)d_in[10];
    const float* bf1    = (const float*)d_in[11];
    const float* Wf2    = (const float*)d_in[12];
    const float* bf2    = (const float*)d_in[13];
    const long long* ei = (const long long*)d_in[14];

    const long long E  = in_sizes[14] / 2;     // edges
    const long long EE = E + NNODES;           // + self loops
    const long long* srcI = ei;
    const long long* dstI = ei + E;

    // ---- workspace layout (region A is reused for packed-z after layer 1) --
    char* ws = (char*)d_ws;
    size_t off = 0;
    auto alloc = [&](size_t bytes) { size_t o = off; off = (off + bytes + 255) & ~(size_t)255; return o; };

    const size_t xpk_sz  = (size_t)MT * 16 * 512 * 2;          // 51.2 MB
    const size_t zpk_sz  = (size_t)MT * 25 * 512 * 2;          // 80.0 MB
    size_t xpk_o   = alloc(xpk_sz);
    size_t w1pk_o  = alloc((size_t)8 * 16 * 512 * 2);
    size_t h1pre_o = alloc((size_t)NNODES * 128 * 4);
    size_t as1_o   = alloc((size_t)NNODES * 4 * 4);
    size_t ad1_o   = alloc((size_t)NNODES * 4 * 4);
    size_t m1_o    = alloc((size_t)NNODES * 4 * 4);
    size_t dn1_o   = alloc((size_t)NNODES * 4 * 4);
    size_t zpk_o   = 0;                                        // aliases region A
    if (off < zpk_sz) off = zpk_sz;
    off = (off + 255) & ~(size_t)255;

    size_t agg1_o  = alloc((size_t)NNODES * 128 * 4);
    size_t h1pk_o  = alloc((size_t)MT * 4 * 512 * 2);
    size_t w2pk_o  = alloc((size_t)2 * 4 * 512 * 2);
    size_t h2pre_o = alloc((size_t)NNODES * 32 * 4);
    size_t as2_o   = alloc((size_t)NNODES * 4);
    size_t ad2_o   = alloc((size_t)NNODES * 4);
    size_t m2_o    = alloc((size_t)NNODES * 4);
    size_t dn2_o   = alloc((size_t)NNODES * 4);
    size_t agg2_o  = alloc((size_t)NNODES * 32 * 4);
    size_t wfpk_o  = alloc((size_t)4 * 25 * 512 * 2);
    size_t z1_o    = alloc((size_t)NNODES * 64 * 4);
    if (ws_size < off) return;                                 // insufficient scratch

    bf16_t* xpk   = (bf16_t*)(ws + xpk_o);
    bf16_t* w1pk  = (bf16_t*)(ws + w1pk_o);
    float*  h1pre = (float*)(ws + h1pre_o);
    float*  as1   = (float*)(ws + as1_o);
    float*  ad1   = (float*)(ws + ad1_o);
    float*  m1    = (float*)(ws + m1_o);
    float*  dn1   = (float*)(ws + dn1_o);
    bf16_t* zpk   = (bf16_t*)(ws + zpk_o);
    float*  agg1  = (float*)(ws + agg1_o);
    bf16_t* h1pk  = (bf16_t*)(ws + h1pk_o);
    bf16_t* w2pk  = (bf16_t*)(ws + w2pk_o);
    float*  h2pre = (float*)(ws + h2pre_o);
    float*  as2   = (float*)(ws + as2_o);
    float*  ad2   = (float*)(ws + ad2_o);
    float*  m2    = (float*)(ws + m2_o);
    float*  dn2   = (float*)(ws + dn2_o);
    float*  agg2  = (float*)(ws + agg2_o);
    bf16_t* wfpk  = (bf16_t*)(ws + wfpk_o);
    float*  z1    = (float*)(ws + z1_o);
    float*  out   = (float*)d_out;

    const float NEGINF = -__builtin_huge_valf();

    // ======================= Layer 1: GAT(512 -> 4x32) ======================
    pack_a_kernel<<<MT * 16, 512, 0, stream>>>(x, xpk, NNODES, 512, 16, 0, nullptr, 0);
    pack_b_kernel<<<8 * 16, 512, 0, stream>>>(W1, w1pk, 512, 128);
    wmma_gemm_kernel<<<MT, 8 * 32, 0, stream>>>(xpk, w1pk, h1pre, 16, 128);

    alpha_kernel<<<(NNODES * 4 + 7) / 8, 256, 0, stream>>>(h1pre, a_src1, a_dst1, as1, ad1, NNODES, 4);

    fill_kernel<<<(NNODES * 4 + 255) / 256, 256, 0, stream>>>(m1, NEGINF, (size_t)NNODES * 4);
    fill_kernel<<<(NNODES * 4 + 255) / 256, 256, 0, stream>>>(dn1, 0.f, (size_t)NNODES * 4);
    fill_kernel<<<((size_t)NNODES * 128 + 255) / 256, 256, 0, stream>>>(agg1, 0.f, (size_t)NNODES * 128);

    edge_max_kernel<<<(unsigned)((EE + 255) / 256), 256, 0, stream>>>(srcI, dstI, E, EE, as1, ad1, m1, 4);
    edge_exp_kernel<<<(unsigned)((EE + 255) / 256), 256, 0, stream>>>(srcI, dstI, E, EE, as1, ad1, m1, dn1, 4);
    edge_agg_kernel<<<(unsigned)((EE + 7) / 8), 256, 0, stream>>>(srcI, dstI, E, EE, as1, ad1, m1, dn1,
                                                                  h1pre, agg1, 4);

    // h1 = elu(agg1 + b1), packed directly as GEMM2 A operand
    pack_a_kernel<<<MT * 4, 512, 0, stream>>>(agg1, h1pk, NNODES, 128, 4, 0, b1, 1);
    // region A now free: pack txt into z (k-tiles 0..23 of 25)
    pack_a_kernel<<<MT * 24, 512, 0, stream>>>(txt, zpk, NNODES, 768, 25, 0, nullptr, 0);

    // ======================= Layer 2: GAT(128 -> 1x32) ======================
    pack_b_kernel<<<2 * 4, 512, 0, stream>>>(W2, w2pk, 128, 32);
    wmma_gemm_kernel<<<MT, 2 * 32, 0, stream>>>(h1pk, w2pk, h2pre, 4, 32);

    alpha_kernel<<<(NNODES + 7) / 8, 256, 0, stream>>>(h2pre, a_src2, a_dst2, as2, ad2, NNODES, 1);

    fill_kernel<<<(NNODES + 255) / 256, 256, 0, stream>>>(m2, NEGINF, (size_t)NNODES);
    fill_kernel<<<(NNODES + 255) / 256, 256, 0, stream>>>(dn2, 0.f, (size_t)NNODES);
    fill_kernel<<<((size_t)NNODES * 32 + 255) / 256, 256, 0, stream>>>(agg2, 0.f, (size_t)NNODES * 32);

    edge_max_kernel<<<(unsigned)((EE + 255) / 256), 256, 0, stream>>>(srcI, dstI, E, EE, as2, ad2, m2, 1);
    edge_exp_kernel<<<(unsigned)((EE + 255) / 256), 256, 0, stream>>>(srcI, dstI, E, EE, as2, ad2, m2, dn2, 1);
    edge_agg_kernel<<<(unsigned)((EE + 7) / 8), 256, 0, stream>>>(srcI, dstI, E, EE, as2, ad2, m2, dn2,
                                                                  h2pre, agg2, 1);

    // g = agg2 + b2, packed into z k-tile 24 (concat [txt | g])
    pack_a_kernel<<<MT * 1, 512, 0, stream>>>(agg2, zpk, NNODES, 32, 25, 24, b2, 0);

    // ============================ Final MLP =================================
    pack_b_kernel<<<4 * 25, 512, 0, stream>>>(Wf1, wfpk, 800, 64);
    wmma_gemm_kernel<<<MT, 4 * 32, 0, stream>>>(zpk, wfpk, z1, 25, 64);

    final_kernel<<<(NNODES + 7) / 8, 256, 0, stream>>>(z1, bf1, Wf2, bf2, out, NNODES);
}
